// SPI_GNN_32452772888694
// MI455X (gfx1250) — matrix operand order repaired and checked by reading, not compiled, not run on hardware
//
#include <hip/hip_runtime.h>
#include <hip/hip_bf16.h>
#include <cmath>

// ---------------------------------------------------------------------------
// Types for CDNA5 WMMA (wave32, 16x16x32 bf16 -> f32)
// ---------------------------------------------------------------------------
typedef __bf16 bf16_t;
typedef bf16_t v16bf __attribute__((ext_vector_type(16)));
typedef bf16_t v8bf  __attribute__((ext_vector_type(8)));
typedef float  v8f   __attribute__((ext_vector_type(8)));

#define SENTF   (-1000000000.0f)
#define ZETAF   (1e-24f)
#define ML      128
#define NPROP   20
#define ENC_NEG_INF 0x007FFFFFu   // enc_order(-inf)

// order-preserving uint encoding of float (for atomic max)
static __device__ __forceinline__ unsigned enc_order(float f) {
  unsigned u = __float_as_uint(f);
  return (u & 0x80000000u) ? ~u : (u | 0x80000000u);
}
static __device__ __forceinline__ float dec_order(unsigned u) {
  unsigned v = (u & 0x80000000u) ? (u & 0x7fffffffu) : ~u;
  return __uint_as_float(v);
}
static __device__ __forceinline__ bool finitef(float f) {
  return (__float_as_uint(f) & 0x7f800000u) != 0x7f800000u;
}

// ---------------------------------------------------------------------------
// WMMA fragment helpers (layouts per CDNA5 ISA 7.12.2, wave32)
// A 16x32 bf16: lane l -> row (l&15); elems 0..7 = K k0..k0+7, 8..15 = K k0+16..k0+23,
//               k0 = kbase + (l>=16 ? 8 : 0)   -> two 16B loads
// ---------------------------------------------------------------------------
static __device__ __forceinline__ v16bf load_frag_a_lds(const bf16_t* base,
                                                        int row_stride, int lane, int kbase) {
  int row = lane & 15;
  int k0  = kbase + ((lane >> 4) << 3);
  union { v16bf v; v8bf h[2]; } u;
  u.h[0] = *(const v8bf*)(base + row * row_stride + k0);
  u.h[1] = *(const v8bf*)(base + row * row_stride + k0 + 16);
  return u.v;
}
// B fragments come from weights pre-packed as [kt][ct][lane][16] (contiguous per lane)
static __device__ __forceinline__ v16bf load_frag_b_pk(const bf16_t* __restrict__ wpk,
                                                       int kt, int ct, int ntn, int lane) {
  const bf16_t* p = wpk + (size_t)((((kt * ntn + ct) << 5) + lane) << 4);
  union { v16bf v; v8bf h[2]; } u;
  u.h[0] = *(const v8bf*)(p);
  u.h[1] = *(const v8bf*)(p + 8);
  return u.v;
}

#define WMMA_BF16(acc, a, b) \
  __builtin_amdgcn_wmma_f32_16x16x32_bf16(false, (a), false, (b), (short)0, (acc), false, false)

// ---------------------------------------------------------------------------
// Weight pre-pack: f32 row-major (fi x fo) -> bf16 fragment-native layout.
// packed[ ((kt*ntn+ct)*32 + lane)*16 + i ] = W[ kt*32 + (lane>>4)*8 + (i&7) + (i>>3)*16 ][ ct*16 + (lane&15) ]
// ---------------------------------------------------------------------------
__global__ void pack_weight_kernel(const float* __restrict__ W, bf16_t* __restrict__ out,
                                   int fi, int fo) {
  int idx = blockIdx.x * blockDim.x + threadIdx.x;
  if (idx >= fi * fo) return;
  int i    = idx & 15;
  int lane = (idx >> 4) & 31;
  int tile = idx >> 9;
  int ntn  = fo >> 4;
  int ct   = tile % ntn;
  int kt   = tile / ntn;
  int K    = kt * 32 + ((lane >> 4) << 3) + (i & 7) + ((i >> 3) << 4);
  int Nc   = ct * 16 + (lane & 15);
  out[idx] = (bf16_t)W[(size_t)K * fo + Nc];
}

__global__ void fill_f32_kernel(float* p, float v, int n) {
  int i = blockIdx.x * blockDim.x + threadIdx.x;
  if (i < n) p[i] = v;
}
__global__ void fill_u32_kernel(unsigned* p, unsigned v, int n) {
  int i = blockIdx.x * blockDim.x + threadIdx.x;
  if (i < n) p[i] = v;
}
__global__ void copy_f32_kernel(const float* __restrict__ s, float* __restrict__ d, int n) {
  int i = blockIdx.x * blockDim.x + threadIdx.x;
  if (i < n) d[i] = s[i];
}

// ---------------------------------------------------------------------------
// g = [d_hat, d_star, res_mask] @ W_node_in   (K=3, scalar FMAs)
// ---------------------------------------------------------------------------
__global__ void node_embed_kernel(const float* __restrict__ dhat, const float* __restrict__ x,
                                  const float* __restrict__ Wni, float* __restrict__ g, int N) {
  int idx = blockIdx.x * blockDim.x + threadIdx.x;
  if (idx >= N * ML) return;
  int n = idx >> 7, c = idx & (ML - 1);
  float v = dhat[n] * Wni[c] + x[(size_t)n * 5 + 1] * Wni[ML + c]
          + x[(size_t)n * 5 + 4] * Wni[2 * ML + c];
  g[idx] = v;
}

// z = [q_tilde, q_hat] @ W_edge  (K=2)
__global__ void edge_embed_kernel(const float* __restrict__ qt, const float* __restrict__ qh,
                                  const float* __restrict__ Wed, float* __restrict__ z, int E) {
  int idx = blockIdx.x * blockDim.x + threadIdx.x;
  if (idx >= E * ML) return;
  int e = idx >> 7, c = idx & (ML - 1);
  z[idx] = qt[e] * Wed[c] + qh[e] * Wed[ML + c];
}

// ---------------------------------------------------------------------------
// Edge MLP layer: z_new = relu([g[s],g[r],z] @ We1) @ We2 ; fused segment_max into agg
// Block = 256 threads (8 waves), 32 edges x 128 cols. K1=384 (12 kt), K2=128 (4 kt).
// ---------------------------------------------------------------------------
__global__ __launch_bounds__(256) void edge_mlp_kernel(
    const float* __restrict__ g, const float* __restrict__ z,
    const int* __restrict__ sndr, const int* __restrict__ rcvr,
    const bf16_t* __restrict__ We1pk, const bf16_t* __restrict__ We2pk,
    float* __restrict__ z_out, unsigned* __restrict__ agg_enc, int E) {
  __shared__ __attribute__((aligned(16))) bf16_t smM[32 * 384];
  __shared__ __attribute__((aligned(16))) bf16_t smH[32 * 128];
  __shared__ int sS[32], sR[32];
  int tid = threadIdx.x;
  int e0  = blockIdx.x * 32;
  if (tid < 32) {
    int e = e0 + tid; if (e >= E) e = E - 1;
    sS[tid] = sndr[e]; sR[tid] = rcvr[e];
  }
  __syncthreads();
  for (int idx = tid; idx < 32 * 384; idx += 256) {
    int row = idx / 384, col = idx - row * 384;
    int e = e0 + row;
    float val;
    if (e >= E)           val = 0.f;
    else if (col < 128)   val = g[(size_t)sS[row] * ML + col];
    else if (col < 256)   val = g[(size_t)sR[row] * ML + (col - 128)];
    else                  val = z[(size_t)e * ML + (col - 256)];
    smM[idx] = (bf16_t)val;
  }
  __syncthreads();
  int wave = tid >> 5, lane = tid & 31;
  // h1 = relu(m @ We1)
  for (int t = 0; t < 2; ++t) {
    int tileid = wave * 2 + t;
    int rt = tileid >> 3, ct = tileid & 7;
    v8f acc = {0.f,0.f,0.f,0.f,0.f,0.f,0.f,0.f};
    for (int kt = 0; kt < 12; ++kt) {
      v16bf a = load_frag_a_lds(&smM[rt * 16 * 384], 384, lane, kt * 32);
      v16bf b = load_frag_b_pk(We1pk, kt, ct, 8, lane);
      acc = WMMA_BF16(acc, a, b);
    }
    int col = ct * 16 + (lane & 15);
    int rb  = rt * 16 + ((lane >> 4) << 3);
#pragma unroll
    for (int v = 0; v < 8; ++v) {
      float f = acc[v]; f = f > 0.f ? f : 0.f;
      smH[(rb + v) * 128 + col] = (bf16_t)f;
    }
  }
  __syncthreads();
  // z_new = h1 @ We2 ; store + atomic segment-max by rcvr
  for (int t = 0; t < 2; ++t) {
    int tileid = wave * 2 + t;
    int rt = tileid >> 3, ct = tileid & 7;
    v8f acc = {0.f,0.f,0.f,0.f,0.f,0.f,0.f,0.f};
    for (int kt = 0; kt < 4; ++kt) {
      v16bf a = load_frag_a_lds(&smH[rt * 16 * 128], 128, lane, kt * 32);
      v16bf b = load_frag_b_pk(We2pk, kt, ct, 8, lane);
      acc = WMMA_BF16(acc, a, b);
    }
    int col = ct * 16 + (lane & 15);
    int rb  = rt * 16 + ((lane >> 4) << 3);
#pragma unroll
    for (int v = 0; v < 8; ++v) {
      int row = rb + v;
      int e = e0 + row;
      if (e < E) {
        float f = acc[v];
        z_out[(size_t)e * ML + col] = f;
        atomicMax(&agg_enc[(size_t)sR[row] * ML + col], enc_order(f));
      }
    }
  }
}

// ---------------------------------------------------------------------------
// Node MLP: g_new = relu([g, where(finite(agg),agg,0)] @ Wn1) @ Wn2
// 32 nodes x 128, K1=256 (8 kt), K2=128 (4 kt)
// ---------------------------------------------------------------------------
__global__ __launch_bounds__(256) void node_mlp_kernel(
    const float* __restrict__ g, const unsigned* __restrict__ agg_enc,
    const bf16_t* __restrict__ Wn1pk, const bf16_t* __restrict__ Wn2pk,
    float* __restrict__ g_out, int N) {
  __shared__ __attribute__((aligned(16))) bf16_t smM[32 * 256];
  __shared__ __attribute__((aligned(16))) bf16_t smH[32 * 128];
  int tid = threadIdx.x;
  int n0  = blockIdx.x * 32;
  for (int idx = tid; idx < 32 * 256; idx += 256) {
    int row = idx >> 8, col = idx & 255;
    int n = n0 + row;
    float val = 0.f;
    if (n < N) {
      if (col < 128) val = g[(size_t)n * ML + col];
      else {
        float f = dec_order(agg_enc[(size_t)n * ML + (col - 128)]);
        val = finitef(f) ? f : 0.f;
      }
    }
    smM[idx] = (bf16_t)val;
  }
  __syncthreads();
  int wave = tid >> 5, lane = tid & 31;
  for (int t = 0; t < 2; ++t) {
    int tileid = wave * 2 + t;
    int rt = tileid >> 3, ct = tileid & 7;
    v8f acc = {0.f,0.f,0.f,0.f,0.f,0.f,0.f,0.f};
    for (int kt = 0; kt < 8; ++kt) {
      v16bf a = load_frag_a_lds(&smM[rt * 16 * 256], 256, lane, kt * 32);
      v16bf b = load_frag_b_pk(Wn1pk, kt, ct, 8, lane);
      acc = WMMA_BF16(acc, a, b);
    }
    int col = ct * 16 + (lane & 15);
    int rb  = rt * 16 + ((lane >> 4) << 3);
#pragma unroll
    for (int v = 0; v < 8; ++v) {
      float f = acc[v]; f = f > 0.f ? f : 0.f;
      smH[(rb + v) * 128 + col] = (bf16_t)f;
    }
  }
  __syncthreads();
  for (int t = 0; t < 2; ++t) {
    int tileid = wave * 2 + t;
    int rt = tileid >> 3, ct = tileid & 7;
    v8f acc = {0.f,0.f,0.f,0.f,0.f,0.f,0.f,0.f};
    for (int kt = 0; kt < 4; ++kt) {
      v16bf a = load_frag_a_lds(&smH[rt * 16 * 128], 128, lane, kt * 32);
      v16bf b = load_frag_b_pk(Wn2pk, kt, ct, 8, lane);
      acc = WMMA_BF16(acc, a, b);
    }
    int col = ct * 16 + (lane & 15);
    int rb  = rt * 16 + ((lane >> 4) << 3);
#pragma unroll
    for (int v = 0; v < 8; ++v) {
      int n = n0 + rb + v;
      if (n < N) g_out[(size_t)n * ML + col] = acc[v];
    }
  }
}

// ---------------------------------------------------------------------------
// z_bar = [g[s], g[r], z] @ W_z_latent   (single K=384 GEMM, no relu)
// ---------------------------------------------------------------------------
__global__ __launch_bounds__(256) void zbar_kernel(
    const float* __restrict__ g, const float* __restrict__ z,
    const int* __restrict__ sndr, const int* __restrict__ rcvr,
    const bf16_t* __restrict__ Wzpk, float* __restrict__ zbar, int E) {
  __shared__ __attribute__((aligned(16))) bf16_t smM[32 * 384];
  __shared__ int sS[32], sR[32];
  int tid = threadIdx.x;
  int e0  = blockIdx.x * 32;
  if (tid < 32) {
    int e = e0 + tid; if (e >= E) e = E - 1;
    sS[tid] = sndr[e]; sR[tid] = rcvr[e];
  }
  __syncthreads();
  for (int idx = tid; idx < 32 * 384; idx += 256) {
    int row = idx / 384, col = idx - row * 384;
    int e = e0 + row;
    float val;
    if (e >= E)         val = 0.f;
    else if (col < 128) val = g[(size_t)sS[row] * ML + col];
    else if (col < 256) val = g[(size_t)sR[row] * ML + (col - 128)];
    else                val = z[(size_t)e * ML + (col - 256)];
    smM[idx] = (bf16_t)val;
  }
  __syncthreads();
  int wave = tid >> 5, lane = tid & 31;
  for (int t = 0; t < 2; ++t) {
    int tileid = wave * 2 + t;
    int rt = tileid >> 3, ct = tileid & 7;
    v8f acc = {0.f,0.f,0.f,0.f,0.f,0.f,0.f,0.f};
    for (int kt = 0; kt < 12; ++kt) {
      v16bf a = load_frag_a_lds(&smM[rt * 16 * 384], 384, lane, kt * 32);
      v16bf b = load_frag_b_pk(Wzpk, kt, ct, 8, lane);
      acc = WMMA_BF16(acc, a, b);
    }
    int col = ct * 16 + (lane & 15);
    int rb  = rt * 16 + ((lane >> 4) << 3);
#pragma unroll
    for (int v = 0; v < 8; ++v) {
      int e = e0 + rb + v;
      if (e < E) zbar[(size_t)e * ML + col] = acc[v];
    }
  }
}

// ---------------------------------------------------------------------------
// q_hat_dir += [zbar[:Eh], zbar[Eh:]] @ W_flows_latent ; pump clamps ; build q_hat
// ---------------------------------------------------------------------------
__global__ void flows_kernel(const float* __restrict__ zbar, const float* __restrict__ Wf,
                             const float* __restrict__ ea, float* __restrict__ qdir,
                             float* __restrict__ qhat, int Eh) {
  int e = blockIdx.x * blockDim.x + threadIdx.x;
  if (e >= Eh) return;
  const float* za = zbar + (size_t)e * ML;
  const float* zb = zbar + (size_t)(e + Eh) * ML;
  float s = 0.f;
#pragma unroll 4
  for (int c = 0; c < ML; ++c) s = fmaf(za[c], Wf[c], s);
#pragma unroll 4
  for (int c = 0; c < ML; ++c) s = fmaf(zb[c], Wf[ML + c], s);
  float q = qdir[e] + s;
  float pd = ea[(size_t)(Eh + e) * 8 + 3];
  if (pd == 1.0f) q = fminf(q, 0.f);
  if (pd == 2.0f) q = 0.f;
  qdir[e]       = q;
  qhat[e]       = q;
  qhat[e + Eh]  = -q;
}

__global__ void segsum_kernel(const float* __restrict__ qh, const int* __restrict__ rcvr,
                              float* __restrict__ dhat, int E) {
  int e = blockIdx.x * blockDim.x + threadIdx.x;
  if (e >= E) return;
  atomicAdd(&dhat[rcvr[e]], qh[e]);
}

// ---------------------------------------------------------------------------
// construct_heads
// ---------------------------------------------------------------------------
__global__ void heads_init_kernel(const float* __restrict__ x, float* __restrict__ h,
                                  int* __restrict__ known, unsigned* __restrict__ hm, int N) {
  int n = blockIdx.x * blockDim.x + threadIdx.x;
  if (n >= N) return;
  float res = x[(size_t)n * 5 + 4];
  bool k = res > 0.5f;
  h[n]     = k ? x[(size_t)n * 5 + 0] : 0.f;
  known[n] = k ? 1 : 0;
  hm[n]    = enc_order(SENTF);
}

__global__ void edge_losses_kernel(const float* __restrict__ qh, const float* __restrict__ ea,
                                   float* __restrict__ l, int E) {
  int e = blockIdx.x * blockDim.x + threadIdx.x;
  if (e >= E) return;
  float q    = qh[e];
  float r    = ea[(size_t)e * 8 + 0];
  float pump = ea[(size_t)e * 8 + 3];
  float c0   = ea[(size_t)e * 8 + 4];
  float c1   = ea[(size_t)e * 8 + 5];
  float c2   = ea[(size_t)e * 8 + 6];
  float q2     = q * q + ZETAF;
  float l_pipe = r * q * powf(q2, 0.426f);
  float l_pump = -(c0 - c1 * powf(q2, c2 * 0.5f));
  l[e] = (pump != 0.f) ? l_pump : l_pipe;
}

__global__ void prop_edge_kernel(const float* __restrict__ h, const int* __restrict__ known,
                                 const float* __restrict__ l, const int* __restrict__ sndr,
                                 const int* __restrict__ rcvr, unsigned* __restrict__ hm, int E) {
  int e = blockIdx.x * blockDim.x + threadIdx.x;
  if (e >= E) return;
  int s = sndr[e];
  if (known[s]) {
    float cand = h[s] - l[e];
    atomicMax(&hm[rcvr[e]], enc_order(cand));
  }
}

__global__ void prop_node_kernel(float* __restrict__ h, int* __restrict__ known,
                                 unsigned* __restrict__ hm, int N) {
  int n = blockIdx.x * blockDim.x + threadIdx.x;
  if (n >= N) return;
  float f = dec_order(hm[n]);
  if (!known[n] && f > SENTF * 0.5f) { h[n] = f; known[n] = 1; }
  hm[n] = enc_order(SENTF);  // reset for next propagation round
}

// q_tilde from heads
__global__ void netflow_kernel(const float* __restrict__ h, const float* __restrict__ ea,
                               const int* __restrict__ sndr, const int* __restrict__ rcvr,
                               float* __restrict__ qt, int E) {
  int e = blockIdx.x * blockDim.x + threadIdx.x;
  if (e >= E) return;
  float dh = h[sndr[e]] - h[rcvr[e]];
  float r  = ea[(size_t)e * 8 + 0];
  float q  = dh * powf(dh * dh + ZETAF, -0.23f) / powf(r, 0.54f);
  float pump = ea[(size_t)e * 8 + 3];
  qt[e] = (pump != 0.f) ? 0.f : q;
}

__global__ void qfix_kernel(const float* __restrict__ ea, const float* __restrict__ qt,
                            float* __restrict__ qh, int E) {
  int e = blockIdx.x * blockDim.x + threadIdx.x;
  if (e >= E) return;
  float pump = ea[(size_t)e * 8 + 3];
  if (fabsf(pump) == 1.0f) qh[e] = qt[e];
}

// ---------------------------------------------------------------------------
// Host orchestration
// ---------------------------------------------------------------------------
static inline int cdiv(int a, int b) { return (a + b - 1) / b; }

extern "C" void kernel_launch(void* const* d_in, const int* in_sizes, int n_in,
                              void* d_out, int out_size, void* d_ws, size_t ws_size,
                              hipStream_t stream) {
  const float* x   = (const float*)d_in[0];
  const float* ea  = (const float*)d_in[1];
  const float* Wni = (const float*)d_in[2];
  const float* Wed = (const float*)d_in[3];
  const float* Wz  = (const float*)d_in[4];
  const float* Wf  = (const float*)d_in[5];
  const float* We1 = (const float*)d_in[6];
  const float* We2 = (const float*)d_in[7];
  const float* Wn1 = (const float*)d_in[8];
  const float* Wn2 = (const float*)d_in[9];
  const int*  eidx = (const int*)d_in[10];
  (void)n_in; (void)out_size; (void)ws_size;

  const int N  = in_sizes[0] / 5;
  const int E  = in_sizes[10] / 2;
  const int Eh = E / 2;
  const int* sndr = eidx;
  const int* rcvr = eidx + E;
  const int K = 3;  // r_iter(1) + n_iter(2): graph-captured fixed structure

  // workspace carve-up
  size_t off = 0;
  auto carve = [&](size_t bytes) -> char* {
    char* p = (char*)d_ws + off;
    off = (off + bytes + 255) & ~(size_t)255;
    return p;
  };
  float*    gA    = (float*)   carve((size_t)N * ML * 4);
  float*    gB    = (float*)   carve((size_t)N * ML * 4);
  float*    zA    = (float*)   carve((size_t)E * ML * 4);
  float*    zB    = (float*)   carve((size_t)E * ML * 4);
  unsigned* agg   = (unsigned*)carve((size_t)N * ML * 4);
  float*    qhat  = (float*)   carve((size_t)E * 4);
  float*    qtil  = (float*)   carve((size_t)E * 4);
  float*    qdir  = (float*)   carve((size_t)Eh * 4);
  float*    dhat  = (float*)   carve((size_t)N * 4);
  float*    hbuf  = (float*)   carve((size_t)N * 4);
  int*      known = (int*)     carve((size_t)N * 4);
  float*    lbuf  = (float*)   carve((size_t)E * 4);
  unsigned* hm    = (unsigned*)carve((size_t)N * 4);
  bf16_t*   we1pk[2] = { (bf16_t*)carve(384 * 128 * 2), (bf16_t*)carve(384 * 128 * 2) };
  bf16_t*   we2pk[2] = { (bf16_t*)carve(128 * 128 * 2), (bf16_t*)carve(128 * 128 * 2) };
  bf16_t*   wn1pk[2] = { (bf16_t*)carve(256 * 128 * 2), (bf16_t*)carve(256 * 128 * 2) };
  bf16_t*   wn2pk[2] = { (bf16_t*)carve(128 * 128 * 2), (bf16_t*)carve(128 * 128 * 2) };
  bf16_t*   wzpk     = (bf16_t*)carve(384 * 128 * 2);

  // pack weights to fragment-native bf16 (re-done every call: deterministic)
  for (int i = 0; i < 2; ++i) {
    pack_weight_kernel<<<cdiv(384 * 128, 256), 256, 0, stream>>>(We1 + (size_t)i * 384 * 128, we1pk[i], 384, 128);
    pack_weight_kernel<<<cdiv(128 * 128, 256), 256, 0, stream>>>(We2 + (size_t)i * 128 * 128, we2pk[i], 128, 128);
    pack_weight_kernel<<<cdiv(256 * 128, 256), 256, 0, stream>>>(Wn1 + (size_t)i * 256 * 128, wn1pk[i], 256, 128);
    pack_weight_kernel<<<cdiv(128 * 128, 256), 256, 0, stream>>>(Wn2 + (size_t)i * 128 * 128, wn2pk[i], 128, 128);
  }
  pack_weight_kernel<<<cdiv(384 * 128, 256), 256, 0, stream>>>(Wz, wzpk, 384, 128);

  // initial state
  fill_f32_kernel<<<cdiv(E, 256), 256, 0, stream>>>(qhat, 0.f, E);
  fill_f32_kernel<<<cdiv(E, 256), 256, 0, stream>>>(qtil, 0.f, E);
  fill_f32_kernel<<<cdiv(Eh, 256), 256, 0, stream>>>(qdir, 0.f, Eh);
  fill_f32_kernel<<<cdiv(N, 256), 256, 0, stream>>>(dhat, 0.f, N);

  float *gcur = gA, *gnxt = gB, *zcur = zA, *znxt = zB;
  const int etiles = cdiv(E, 32), ntiles = cdiv(N, 32);

  for (int k = 0; k < K; ++k) {
    node_embed_kernel<<<cdiv(N * ML, 256), 256, 0, stream>>>(dhat, x, Wni, gcur, N);
    edge_embed_kernel<<<cdiv(E * ML, 256), 256, 0, stream>>>(qtil, qhat, Wed, zcur, E);

    for (int i = 0; i < 2; ++i) {
      fill_u32_kernel<<<cdiv(N * ML, 256), 256, 0, stream>>>(agg, ENC_NEG_INF, N * ML);
      edge_mlp_kernel<<<etiles, 256, 0, stream>>>(gcur, zcur, sndr, rcvr,
                                                  we1pk[i], we2pk[i], znxt, agg, E);
      node_mlp_kernel<<<ntiles, 256, 0, stream>>>(gcur, agg, wn1pk[i], wn2pk[i], gnxt, N);
      float* t;
      t = gcur; gcur = gnxt; gnxt = t;
      t = zcur; zcur = znxt; znxt = t;
    }

    // z_bar into the free z buffer, then flow update
    zbar_kernel<<<etiles, 256, 0, stream>>>(gcur, zcur, sndr, rcvr, wzpk, znxt, E);
    flows_kernel<<<cdiv(Eh, 256), 256, 0, stream>>>(znxt, Wf, ea, qdir, qhat, Eh);
    // (reference's first segment_sum of d_hat is dead: overwritten below before use)

    // construct_heads: 20 rounds of relax + commit
    heads_init_kernel<<<cdiv(N, 256), 256, 0, stream>>>(x, hbuf, known, hm, N);
    edge_losses_kernel<<<cdiv(E, 256), 256, 0, stream>>>(qhat, ea, lbuf, E);
    for (int p = 0; p < NPROP; ++p) {
      prop_edge_kernel<<<cdiv(E, 256), 256, 0, stream>>>(hbuf, known, lbuf, sndr, rcvr, hm, E);
      prop_node_kernel<<<cdiv(N, 256), 256, 0, stream>>>(hbuf, known, hm, N);
    }

    netflow_kernel<<<cdiv(E, 256), 256, 0, stream>>>(hbuf, ea, sndr, rcvr, qtil, E);
    qfix_kernel<<<cdiv(E, 256), 256, 0, stream>>>(ea, qtil, qhat, E);
    fill_f32_kernel<<<cdiv(N, 256), 256, 0, stream>>>(dhat, 0.f, N);
    segsum_kernel<<<cdiv(E, 256), 256, 0, stream>>>(qhat, rcvr, dhat, E);
  }

  copy_f32_kernel<<<cdiv(N, 256), 256, 0, stream>>>(hbuf, (float*)d_out, N);
}
